// Attention_33011118637863
// MI455X (gfx1250) — matrix operand order repaired and checked by reading, not compiled, not run on hardware
//
#include <hip/hip_runtime.h>

typedef __attribute__((ext_vector_type(16))) __bf16 v16bf;
typedef __attribute__((ext_vector_type(8)))  __bf16 v8bf;
typedef __attribute__((ext_vector_type(4)))  __bf16 v4bf;
typedef __attribute__((ext_vector_type(8)))  float  v8f;
typedef __attribute__((ext_vector_type(4)))  float  v4f;

#define WMMA_BF16(A, B, C) \
  __builtin_amdgcn_wmma_f32_16x16x32_bf16(false, (A), false, (B), (short)0, (C), false, false)

__device__ __forceinline__ float bf2f(__bf16 x) { return (float)x; }
__device__ __forceinline__ __bf16 f2bf(float x) { return (__bf16)x; }

// Build a 16-element bf16 fragment from two 8-element (16B) chunks.
__device__ __forceinline__ v16bf load_frag(const __bf16* lo, const __bf16* hi) {
  v8bf a = *(const v8bf*)lo;
  v8bf b = *(const v8bf*)hi;
  v16bf r;
#pragma unroll
  for (int i = 0; i < 8; ++i) { r[i] = a[i]; r[i + 8] = b[i]; }
  return r;
}

// ---------------------------------------------------------------- fp32 -> bf16 (x4 vectorized)
__global__ void k_cvt_bf16x4(const float* __restrict__ s, __bf16* __restrict__ d, int n4) {
  int i = blockIdx.x * 256 + threadIdx.x;
  if (i < n4) {
    v4f v = *(const v4f*)(s + (size_t)i * 4);
    v4bf o;
#pragma unroll
    for (int j = 0; j < 4; ++j) o[j] = f2bf(v[j]);
    *(v4bf*)(d + (size_t)i * 4) = o;
  }
}

// ---------------------------------------------------------------- QKV GEMM
// x_bf16 [8192 x 768] @ qkv_w_bf16^T [768 x 2304] + qkv_b
// q -> q_bf16 (bh,1024,64); k -> pre_kv[0] f32 + k_bf16; v -> pre_kv[1] f32 + vt_bf16 (bh,64,1024)
__global__ __launch_bounds__(256) void k_qkv_gemm(
    const __bf16* __restrict__ xb, const __bf16* __restrict__ wb,
    const float* __restrict__ bias,
    __bf16* __restrict__ qb, __bf16* __restrict__ kb, __bf16* __restrict__ vtb,
    float* __restrict__ prekv) {
  const int lane = threadIdx.x & 31, wid = threadIdx.x >> 5;
  const int lr = lane & 15, half = lane >> 4;
  const int koffA = half * 8, koffB = half * 16;
  const int m0 = blockIdx.y * 128 + (wid & 3) * 32;
  const int n0 = blockIdx.x * 128 + (wid >> 2) * 64;

  v8f acc[2][4];
#pragma unroll
  for (int i = 0; i < 2; ++i)
#pragma unroll
    for (int j = 0; j < 4; ++j) acc[i][j] = (v8f)0.0f;

#pragma unroll 2
  for (int kk = 0; kk < 768; kk += 32) {
    v16bf a[2];
#pragma unroll
    for (int i = 0; i < 2; ++i) {
      const __bf16* base = xb + (size_t)(m0 + i * 16 + lr) * 768 + kk;
      a[i] = load_frag(base + koffB, base + koffB + 8);
    }
#pragma unroll
    for (int j = 0; j < 4; ++j) {
      const __bf16* base = wb + (size_t)(n0 + j * 16 + lr) * 768 + kk;
      v16bf bf = load_frag(base + koffB, base + koffB + 8);
#pragma unroll
      for (int i = 0; i < 2; ++i) acc[i][j] = WMMA_BF16(a[i], bf, acc[i][j]);
    }
  }

#pragma unroll
  for (int i = 0; i < 2; ++i)
#pragma unroll
    for (int j = 0; j < 4; ++j) {
      const int n = n0 + j * 16 + lr;
      const float bv = bias[n];
#pragma unroll
      for (int r = 0; r < 8; ++r) {
        const int m = m0 + i * 16 + r + 8 * half;
        const float v = acc[i][j][r] + bv;
        const int bidx = m >> 10, tok = m & 1023;
        if (n < 768) {
          const int head = n >> 6, d = n & 63;
          const int bh = bidx * 12 + head;
          qb[(size_t)(bh * 1024 + tok) * 64 + d] = f2bf(v);
        } else if (n < 1536) {
          const int c = n - 768;
          const int head = c >> 6, d = c & 63;
          const int bh = bidx * 12 + head;
          const size_t idx = (size_t)(bh * 1024 + tok) * 64 + d;
          prekv[idx] = v;
          kb[idx] = f2bf(v);
        } else {
          const int c = n - 1536;
          const int head = c >> 6, d = c & 63;
          const int bh = bidx * 12 + head;
          prekv[(size_t)6291456 + (size_t)(bh * 1024 + tok) * 64 + d] = v;
          vtb[(size_t)(bh * 64 + d) * 1024 + tok] = f2bf(v);
        }
      }
    }
}

// ---------------------------------------------------------------- rel-pos bias
// rel_h[bh,qi,kh] = sum_c q[bh,qi,c] * rel_pos_h[h-kh+31,c]   (h = qi>>5)
// rel_w[bh,qi,kw] = sum_c q[bh,qi,c] * rel_pos_w[w-kw+31,c]   (w = qi&31)
__global__ void k_relpos(const __bf16* __restrict__ qb,
                         const float* __restrict__ rph, const float* __restrict__ rpw,
                         float* __restrict__ relh, float* __restrict__ relw) {
  const int idx = blockIdx.x * 256 + threadIdx.x;  // [0, 96*1024*32)
  const int kk = idx & 31;
  const int qi = (idx >> 5) & 1023;
  const int bh = idx >> 15;
  const __bf16* q = qb + (size_t)(bh * 1024 + qi) * 64;
  const int h = qi >> 5, w = qi & 31;
  const float* th = rph + (size_t)(h - kk + 31) * 64;
  const float* tw = rpw + (size_t)(w - kk + 31) * 64;
  float sh = 0.f, sw = 0.f;
#pragma unroll 8
  for (int c = 0; c < 64; ++c) {
    const float qv = bf2f(q[c]);
    sh += qv * th[c];
    sw += qv * tw[c];
  }
  relh[idx] = sh;
  relw[idx] = sw;
}

// ---------------------------------------------------------------- flash attention
// One block per (128-row q-tile, batch-head). 8 waves, wave w owns 16 q rows.
// All LDS use is wave-private: NO block barriers anywhere in this kernel.
__global__ __launch_bounds__(256) void k_flash_attn(
    const __bf16* __restrict__ qb, const __bf16* __restrict__ kb,
    const __bf16* __restrict__ vtb,
    const float* __restrict__ relh, const float* __restrict__ relw,
    __bf16* __restrict__ attnb) {
  __shared__ float sh_rh[128 * 32];
  __shared__ float sh_rw[128 * 32];
  __shared__ __align__(16) __bf16 sh_p[8 * 16 * 64];

  const int bh = blockIdx.y;
  const int q0 = blockIdx.x * 128;
  const int b = bh / 12, head = bh % 12;
  const int tid = threadIdx.x;
  const int lane = tid & 31, wid = tid >> 5;
  const int lr = lane & 15, half = lane >> 4;
  const int koffA = half * 8, koffB = half * 16;

  // Each wave stages the rel-pos rows for ITS OWN 16 query rows -> no barrier needed.
  const int row0 = wid * 16;
  for (int i = lane; i < 16 * 32; i += 32) {
    const int rr = row0 + (i >> 5), cc = i & 31;
    const size_t g = (size_t)(bh * 1024 + q0 + rr) * 32 + cc;
    sh_rh[rr * 32 + cc] = relh[g];
    sh_rw[rr * 32 + cc] = relw[g];
  }

  const int qrow = q0 + row0;
  v16bf aq[2];
#pragma unroll
  for (int kk = 0; kk < 2; ++kk) {
    const __bf16* base = qb + (size_t)(bh * 1024 + qrow + lr) * 64 + kk * 32;
    aq[kk] = load_frag(base + koffB, base + koffB + 8);
  }

  v8f oacc[4];
#pragma unroll
  for (int t = 0; t < 4; ++t) oacc[t] = (v8f)0.0f;
  float mx[8], lsum[8];
#pragma unroll
  for (int r = 0; r < 8; ++r) { mx[r] = -1e30f; lsum[r] = 0.f; }

  for (int kt = 0; kt < 1024; kt += 64) {
    // S = q @ k^T (16 x 64 per wave)
    v8f s[4];
#pragma unroll
    for (int t = 0; t < 4; ++t) s[t] = (v8f)0.0f;
#pragma unroll
    for (int t = 0; t < 4; ++t) {
      const __bf16* kbase = kb + (size_t)(bh * 1024 + kt + t * 16 + lr) * 64;
#pragma unroll
      for (int kk = 0; kk < 2; ++kk) {
        v16bf bf = load_frag(kbase + kk * 32 + koffB, kbase + kk * 32 + koffB + 8);
        s[t] = WMMA_BF16(aq[kk], bf, s[t]);
      }
    }
    // scale + decomposed rel-pos bias
#pragma unroll
    for (int t = 0; t < 4; ++t) {
      const int col = t * 16 + lr;
      const int kj = kt + col;
      const int khidx = kj >> 5, kwidx = kj & 31;
#pragma unroll
      for (int r = 0; r < 8; ++r) {
        const int ml = row0 + r + 8 * half;
        s[t][r] = s[t][r] * 0.125f + sh_rh[ml * 32 + khidx] + sh_rw[ml * 32 + kwidx];
      }
    }
    // online softmax: row stats replicated across each 16-lane half
    float alpha[8];
#pragma unroll
    for (int r = 0; r < 8; ++r) {
      float rm = fmaxf(fmaxf(s[0][r], s[1][r]), fmaxf(s[2][r], s[3][r]));
#pragma unroll
      for (int msk = 1; msk < 16; msk <<= 1) rm = fmaxf(rm, __shfl_xor(rm, msk, 32));
      const float mnew = fmaxf(mx[r], rm);
      alpha[r] = __expf(mx[r] - mnew);
      mx[r] = mnew;
    }
#pragma unroll
    for (int r = 0; r < 8; ++r) {
      float ps = 0.f;
#pragma unroll
      for (int t = 0; t < 4; ++t) {
        const float p = __expf(s[t][r] - mx[r]);
        s[t][r] = p;
        ps += p;
      }
#pragma unroll
      for (int msk = 1; msk < 16; msk <<= 1) ps += __shfl_xor(ps, msk, 32);
      lsum[r] = lsum[r] * alpha[r] + ps;
#pragma unroll
      for (int t = 0; t < 4; ++t) oacc[t][r] *= alpha[r];
    }
    // stage P (bf16) into this wave's private LDS region, re-read in A-fragment layout.
    // Same-wave LDS ordering is guaranteed (in-order LDS + DScnt waits).
    __bf16* pw = sh_p + wid * 1024;
#pragma unroll
    for (int t = 0; t < 4; ++t)
#pragma unroll
      for (int r = 0; r < 8; ++r)
        pw[(r + 8 * half) * 64 + t * 16 + lr] = f2bf(s[t][r]);
    // O += P @ V  (V transposed: vt[bh, d, n])
#pragma unroll
    for (int kk = 0; kk < 2; ++kk) {
      const __bf16* pbase = sh_p + wid * 1024 + lr * 64 + kk * 32;
      v16bf ap = load_frag(pbase + koffB, pbase + koffB + 8);
#pragma unroll
      for (int t = 0; t < 4; ++t) {
        const __bf16* vbase = vtb + (size_t)(bh * 64 + t * 16 + lr) * 1024 + kt + kk * 32;
        v16bf bv = load_frag(vbase + koffB, vbase + koffB + 8);
        oacc[t] = WMMA_BF16(ap, bv, oacc[t]);
      }
    }
  }

  // normalize and write to (B, N, C) bf16 for the proj GEMM
#pragma unroll
  for (int t = 0; t < 4; ++t)
#pragma unroll
    for (int r = 0; r < 8; ++r) {
      const int tok = qrow + r + 8 * half;
      const int d = t * 16 + lr;
      const float v = oacc[t][r] / lsum[r];
      attnb[(size_t)(b * 1024 + tok) * 768 + head * 64 + d] = f2bf(v);
    }
}

// ---------------------------------------------------------------- proj GEMM
// attn_bf16 [8192 x 768] @ proj_w_bf16^T [768 x 768] + proj_b -> out f32
__global__ __launch_bounds__(256) void k_proj_gemm(
    const __bf16* __restrict__ ab, const __bf16* __restrict__ wb,
    const float* __restrict__ bias, float* __restrict__ out) {
  const int lane = threadIdx.x & 31, wid = threadIdx.x >> 5;
  const int lr = lane & 15, half = lane >> 4;
  const int koffA = half * 8, koffB = half * 16;
  const int m0 = blockIdx.y * 128 + (wid & 3) * 32;
  const int n0 = blockIdx.x * 128 + (wid >> 2) * 64;

  v8f acc[2][4];
#pragma unroll
  for (int i = 0; i < 2; ++i)
#pragma unroll
    for (int j = 0; j < 4; ++j) acc[i][j] = (v8f)0.0f;

#pragma unroll 2
  for (int kk = 0; kk < 768; kk += 32) {
    v16bf a[2];
#pragma unroll
    for (int i = 0; i < 2; ++i) {
      const __bf16* base = ab + (size_t)(m0 + i * 16 + lr) * 768 + kk;
      a[i] = load_frag(base + koffB, base + koffB + 8);
    }
#pragma unroll
    for (int j = 0; j < 4; ++j) {
      const __bf16* base = wb + (size_t)(n0 + j * 16 + lr) * 768 + kk;
      v16bf bf = load_frag(base + koffB, base + koffB + 8);
#pragma unroll
      for (int i = 0; i < 2; ++i) acc[i][j] = WMMA_BF16(a[i], bf, acc[i][j]);
    }
  }

#pragma unroll
  for (int i = 0; i < 2; ++i)
#pragma unroll
    for (int j = 0; j < 4; ++j) {
      const int n = n0 + j * 16 + lr;
      const float bv = bias[n];
#pragma unroll
      for (int r = 0; r < 8; ++r) {
        const int m = m0 + i * 16 + r + 8 * half;
        out[(size_t)m * 768 + n] = acc[i][j][r] + bv;
      }
    }
}

extern "C" void kernel_launch(void* const* d_in, const int* in_sizes, int n_in,
                              void* d_out, int out_size, void* d_ws, size_t ws_size,
                              hipStream_t stream) {
  (void)in_sizes; (void)n_in; (void)out_size; (void)ws_size;
  const float* x      = (const float*)d_in[0];
  const float* qkv_w  = (const float*)d_in[1];
  const float* qkv_b  = (const float*)d_in[2];
  const float* proj_w = (const float*)d_in[3];
  const float* proj_b = (const float*)d_in[4];
  const float* rph    = (const float*)d_in[5];
  const float* rpw    = (const float*)d_in[6];
  // d_in[7], d_in[8]: H=W=32 (hardcoded)

  float* out   = (float*)d_out;
  float* prekv = out + 6291456;  // 8*1024*768

  char* ws = (char*)d_ws;
  size_t off = 0;
  auto alloc = [&](size_t bytes) -> void* {
    void* p = ws + off;
    off = (off + bytes + 255) & ~(size_t)255;
    return p;
  };
  __bf16* xb     = (__bf16*)alloc((size_t)8192 * 768 * 2);
  __bf16* wqkvb  = (__bf16*)alloc((size_t)2304 * 768 * 2);
  __bf16* wprojb = (__bf16*)alloc((size_t)768 * 768 * 2);
  __bf16* qb     = (__bf16*)alloc((size_t)96 * 1024 * 64 * 2);
  __bf16* kbf    = (__bf16*)alloc((size_t)96 * 1024 * 64 * 2);
  __bf16* vtb    = (__bf16*)alloc((size_t)96 * 64 * 1024 * 2);
  float*  relh   = (float*)alloc((size_t)96 * 1024 * 32 * 4);
  float*  relw   = (float*)alloc((size_t)96 * 1024 * 32 * 4);
  __bf16* attnb  = (__bf16*)alloc((size_t)8192 * 768 * 2);

  k_cvt_bf16x4<<<(6291456 / 4 + 255) / 256, 256, 0, stream>>>(x, xb, 6291456 / 4);
  k_cvt_bf16x4<<<(1769472 / 4 + 255) / 256, 256, 0, stream>>>(qkv_w, wqkvb, 1769472 / 4);
  k_cvt_bf16x4<<<(589824 / 4 + 255) / 256, 256, 0, stream>>>(proj_w, wprojb, 589824 / 4);
  k_qkv_gemm<<<dim3(18, 64), 256, 0, stream>>>(xb, wqkvb, qkv_b, qb, kbf, vtb, prekv);
  k_relpos<<<3145728 / 256, 256, 0, stream>>>(qb, rph, rpw, relh, relw);
  k_flash_attn<<<dim3(8, 96), 256, 0, stream>>>(qb, kbf, vtb, relh, relw, attnb);
  k_proj_gemm<<<dim3(6, 64), 256, 0, stream>>>(attnb, wprojb, proj_b, out);
}
